// Deepseekv2Attention_82429012345607
// MI455X (gfx1250) — matrix operand order repaired and checked by reading, compiled, not run on hardware
//
#include <hip/hip_runtime.h>
#include <cmath>

// ---------------------------------------------------------------------------
// DeepSeek-V2 MLA attention for MI455X (gfx1250, wave32, WMMA).
// All contractions via v_wmma_f32_16x16x32_bf16 (bf16 in, f32 acc).
// Fused flash-style MLA core; K-tile staging via TDM (tensor_load_to_lds,
// TENSORcnt) or global_load_async_to_lds_b128 (ASYNCcnt), double-buffered.
// ---------------------------------------------------------------------------

typedef __bf16 bf16;
typedef __attribute__((ext_vector_type(16))) __bf16 v16bf;
typedef __attribute__((ext_vector_type(8)))  __bf16 v8bf;
typedef __attribute__((ext_vector_type(8)))  float  v8f;

constexpr int kB    = 2;
constexpr int kS    = 2048;
constexpr int kHID  = 2048;
constexpr int kNH   = 16;
constexpr int kHD   = 128;
constexpr int kQHD  = 256;
constexpr int kQLR  = 1536;
constexpr int kKVLR = 512;

// --- CDNA5 feature detection (device pass only) -----------------------------
#if defined(__AMDGCN__)
  #if __has_builtin(__builtin_amdgcn_global_load_async_to_lds_b128) && \
      __has_builtin(__builtin_amdgcn_s_wait_asynccnt)
    #define USE_ASYNC 1
  #else
    #define USE_ASYNC 0
  #endif
  #if __has_builtin(__builtin_amdgcn_tensor_load_to_lds) && \
      __has_builtin(__builtin_amdgcn_s_wait_tensorcnt)
    #define USE_TDM 1
  #else
    #define USE_TDM 0
  #endif
#else
  #define USE_ASYNC 0
  #define USE_TDM 0
#endif

#if USE_ASYNC
  #define ASYNC_WAIT(n) __builtin_amdgcn_s_wait_asynccnt(n)
#else
  #define ASYNC_WAIT(n) ((void)0)
#endif
#if USE_TDM
  #define TENSOR_WAIT(n) __builtin_amdgcn_s_wait_tensorcnt(n)
#else
  #define TENSOR_WAIT(n) ((void)0)
#endif

// 16B global -> LDS copy: async-to-LDS when available (ASYNCcnt), else sync.
#if USE_ASYNC
typedef int v4i_vs __attribute__((vector_size(16)));
__device__ inline void g2l_b128(const bf16* g, bf16* l) {
  // builtin signature: (v4i32 addrspace(1)*, v4i32 addrspace(3)*, imm off, imm cpol)
  __builtin_amdgcn_global_load_async_to_lds_b128(
      (__attribute__((address_space(1))) v4i_vs*)(void*)(const_cast<bf16*>(g)),
      (__attribute__((address_space(3))) v4i_vs*)(void*)l, 0, 0);
}
#else
__device__ inline void g2l_b128(const bf16* g, bf16* l) {
  *(v8bf*)l = *(const v8bf*)g;
}
#endif

#if USE_TDM
typedef unsigned int u32x4 __attribute__((vector_size(16)));
typedef int          i32x8 __attribute__((vector_size(32)));
typedef int          i32x4 __attribute__((vector_size(16)));

__device__ inline unsigned lds_byte_off(void* p) {
  return (unsigned)(unsigned long long)(__attribute__((address_space(3))) char*)p;
}

// 1-D contiguous TDM copy: nelem bf16 elements, global -> LDS.
// D# per CDNA5 ISA ch.8: group0 = {count=1, lds_addr, global_addr, type=2};
// group1 = {data_size=2B, tensor_dim0=tile_dim0=nelem, tensor_dim1=1, stride}.
__device__ inline void tdm_copy_1d(const void* gsrc, void* ldst, unsigned nelem) {
  unsigned long long ga = (unsigned long long)gsrc;
  unsigned la = lds_byte_off(ldst);
  u32x4 g0;
  g0[0] = 1u;                                             // count=1, user mode
  g0[1] = la;                                             // lds_addr [63:32]
  g0[2] = (unsigned)(ga & 0xFFFFFFFFu);                   // global_addr lo
  g0[3] = (unsigned)((ga >> 32) & 0x01FFFFFFu) | (2u << 30); // addr hi | type=2
  i32x8 g1;
  g1[0] = (int)(1u << 16);                                // data_size=1 (2B)
  g1[1] = (int)((nelem & 0xFFFFu) << 16);                 // tensor_dim0[15:0]
  g1[2] = (int)(((nelem >> 16) & 0xFFFFu) | (1u << 16));  // td0[31:16] | td1=1
  g1[3] = (int)((nelem & 0xFFFFu) << 16);                 // tile_dim0 = nelem
  g1[4] = 0;                                              // tile_dim1/2 = 0
  g1[5] = (int)nelem;                                     // dim0_stride lo
  g1[6] = 0;
  g1[7] = 0;
  i32x4 z4 = {0, 0, 0, 0};
  i32x8 z8 = {0, 0, 0, 0, 0, 0, 0, 0};
  // 6-arg form (clang-23 / therock): (g0, g1, g2, g3, g4, cpol)
  __builtin_amdgcn_tensor_load_to_lds(g0, g1, z4, z4, z8, 0);
}
#endif

union Frag16 { v16bf v; v8bf h[2]; bf16 e[16]; };

// --- WMMA fragment loaders (contiguous k = 16*(lane/16) + s in A and B, wave32) -----------
__device__ inline v16bf lds_a_frag(const bf16* lds, int row_base, int ld, int k0) {
  int lane = threadIdx.x & 31;
  int m = lane & 15, g = lane >> 4;
  const bf16* p = lds + (row_base + m) * ld + k0 + g * 16;
  Frag16 u;
  u.h[0] = *(const v8bf*)p;
  u.h[1] = *(const v8bf*)(p + 8);
  return u.v;
}
__device__ inline v16bf lds_b_frag(const bf16* lds, int n_base, int ld, int k0) {
  int lane = threadIdx.x & 31;
  int n = lane & 15, g = lane >> 4;
  const bf16* p = lds + (n_base + n) * ld + k0 + g * 16;
  Frag16 u;
  u.h[0] = *(const v8bf*)p;
  u.h[1] = *(const v8bf*)(p + 8);
  return u.v;
}
// B fragment from a [k][n]-stored LDS tile (transposed gather).
__device__ inline v16bf lds_bt_frag(const bf16* lds, int k_base, int n_base, int ld) {
  int lane = threadIdx.x & 31;
  int n = lane & 15, g = lane >> 4;
  Frag16 u;
#pragma unroll
  for (int i = 0; i < 16; ++i)
    u.e[i] = lds[(k_base + g * 16 + i) * ld + n_base + n];
  return u.v;
}

__device__ inline v8f wmma_bf16(v16bf a, v16bf b, v8f c) {
  return __builtin_amdgcn_wmma_f32_16x16x32_bf16(false, a, false, b, (short)0, c,
                                                 false, false);
}

// ---------------------------------------------------------------------------
// Elementwise f32 -> bf16 cast
// ---------------------------------------------------------------------------
__global__ __launch_bounds__(256)
void cast_f32_bf16_kernel(const float* __restrict__ in, bf16* __restrict__ out, long n) {
  long i = (long)blockIdx.x * blockDim.x + threadIdx.x;
  long stride = (long)gridDim.x * blockDim.x;
  for (; i < n; i += stride) out[i] = (bf16)in[i];
}

// ---------------------------------------------------------------------------
// Generic strided/batched bf16 WMMA GEMM, double-buffered async staging.
// C[z] = A[z] @ B[z](^T via strides) + bias.  Tile 128x128x32, 8 waves.
// ---------------------------------------------------------------------------
__global__ __launch_bounds__(256)
void gemm_bf16_kernel(const bf16* __restrict__ A, const bf16* __restrict__ Bm,
                      const float* __restrict__ bias,
                      float* __restrict__ Cf, bf16* __restrict__ Cb,
                      int M, int N, int K,
                      long ldA, long sBN, long sBK, long ldC,
                      int zDiv,
                      long aOut, long aIn, long bOut, long bIn, long cOut, long cIn) {
  __shared__ __align__(16) bf16 sA[2][128 * 32];
  __shared__ __align__(16) bf16 sB[2][128 * 32];

  int z = blockIdx.z;
  A  += (long)(z / zDiv) * aOut + (long)(z % zDiv) * aIn;
  Bm += (long)(z / zDiv) * bOut + (long)(z % zDiv) * bIn;
  long cOff = (long)(z / zDiv) * cOut + (long)(z % zDiv) * cIn;

  int tid  = threadIdx.x;
  int lane = tid & 31;
  int w    = tid >> 5;
  int wm   = w >> 1;   // rows wm*32
  int wn   = w & 1;    // cols wn*64
  int m0   = blockIdx.y * 128;
  int n0   = blockIdx.x * 128;

  v8f acc[2][4];
#pragma unroll
  for (int i = 0; i < 2; ++i)
#pragma unroll
    for (int j = 0; j < 4; ++j)
#pragma unroll
      for (int v = 0; v < 8; ++v) acc[i][j][v] = 0.f;

  auto stageAB = [&](int kt, int bsel) {
    int k0 = kt * 32;
#pragma unroll
    for (int it = 0; it < 2; ++it) {
      int vi = tid + it * 256;
      int r = vi >> 2, c = (vi & 3) * 8;
      g2l_b128(A + (long)(m0 + r) * ldA + k0 + c, &sA[bsel][r * 32 + c]);
    }
    if (sBK == 1) {       // K-contiguous B: vectorized async
#pragma unroll
      for (int it = 0; it < 2; ++it) {
        int vi = tid + it * 256;
        int r = vi >> 2, c = (vi & 3) * 8;
        g2l_b128(Bm + (long)(n0 + r) * sBN + k0 + c, &sB[bsel][r * 32 + c]);
      }
    } else {              // strided-K B (o_absorb path): scalar gathers
      for (int e = tid; e < 128 * 32; e += 256) {
        int r = e >> 5, c = e & 31;
        const bf16* gp = Bm + (long)(n0 + r) * sBN + (long)(k0 + c) * sBK;
        sB[bsel][r * 32 + c] = *gp;
        if (k0 + 32 < K) __builtin_prefetch(gp + 32 * sBK, 0, 0);
      }
    }
  };

  int ktiles = K >> 5;
  int buf = 0;
  stageAB(0, 0);
  for (int kt = 0; kt < ktiles; ++kt) {
    bool hn = (kt + 1 < ktiles);
    if (hn) stageAB(kt + 1, buf ^ 1);
    // ASYNCcnt: vector path issues 4 per thread per tile, gather path 2.
    if (sBK == 1) { if (hn) ASYNC_WAIT(4); else ASYNC_WAIT(0); }
    else          { if (hn) ASYNC_WAIT(2); else ASYNC_WAIT(0); }
    __syncthreads();

    v16bf a0 = lds_a_frag(&sA[buf][0], wm * 32,      32, 0);
    v16bf a1 = lds_a_frag(&sA[buf][0], wm * 32 + 16, 32, 0);
#pragma unroll
    for (int cb = 0; cb < 4; ++cb) {
      v16bf bfr = lds_b_frag(&sB[buf][0], wn * 64 + cb * 16, 32, 0);
      acc[0][cb] = wmma_bf16(a0, bfr, acc[0][cb]);
      acc[1][cb] = wmma_bf16(a1, bfr, acc[1][cb]);
    }
    __syncthreads();
    buf ^= 1;
  }

  int g = lane >> 4, nn = lane & 15;
#pragma unroll
  for (int rb = 0; rb < 2; ++rb)
#pragma unroll
    for (int cb = 0; cb < 4; ++cb) {
      int col = n0 + wn * 64 + cb * 16 + nn;
      float bv = bias ? bias[col] : 0.f;
#pragma unroll
      for (int v = 0; v < 8; ++v) {
        int row = m0 + wm * 32 + rb * 16 + v + 8 * g;
        if (row < M && col < N) {
          float val = acc[rb][cb][v] + bv;
          long idx = cOff + (long)row * ldC + col;
          if (Cf) Cf[idx] = val;
          if (Cb) Cb[idx] = (bf16)val;
        }
      }
    }
}

// ---------------------------------------------------------------------------
// RMSNorm (reference: rsqrt(sum(x*x) + eps), no mean) -> bf16
// ---------------------------------------------------------------------------
__global__ __launch_bounds__(256)
void rmsnorm_kernel(const float* __restrict__ x, const float* __restrict__ w,
                    bf16* __restrict__ out, int cols) {
  long row = blockIdx.x;
  const float* xr = x + row * cols;
  float s = 0.f;
  for (int i = threadIdx.x; i < cols; i += 256) { float v = xr[i]; s += v * v; }
  __shared__ float red[256];
  red[threadIdx.x] = s;
  __syncthreads();
  for (int k = 128; k > 0; k >>= 1) {
    if (threadIdx.x < k) red[threadIdx.x] += red[threadIdx.x + k];
    __syncthreads();
  }
  float inv = rsqrtf(red[0] + 1e-6f);
  for (int i = threadIdx.x; i < cols; i += 256)
    out[row * cols + i] = (bf16)(xr[i] * inv * w[i]);
}

// ---------------------------------------------------------------------------
// RoPE + layout split for q
// ---------------------------------------------------------------------------
__global__ __launch_bounds__(256)
void rope_q_kernel(const float* __restrict__ q, bf16* __restrict__ q_nope,
                   bf16* __restrict__ q_pe) {
  int row = blockIdx.x;            // b*S + s
  int b = row / kS, s = row % kS;
  const float* qr = q + (long)row * (kNH * kQHD);
  for (int e = threadIdx.x; e < kNH * kHD; e += 256) {
    int h = e >> 7, d = e & 127;
    long o = ((long)(b * kNH + h) * kS + s) * kHD + d;
    q_nope[o] = (bf16)qr[h * kQHD + d];
    float x = qr[h * kQHD + kHD + d];
    int dm = d & 63;
    float invf = powf(10000.0f, -(float)(2 * dm) / 128.0f);
    float ang = (float)s * invf;
    float x2 = (d < 64) ? -qr[h * kQHD + kHD + d + 64]
                        :  qr[h * kQHD + kHD + d - 64];
    q_pe[o] = (bf16)(x * cosf(ang) + x2 * sinf(ang));
  }
}

__global__ __launch_bounds__(256)
void rope_kv_kernel(const float* __restrict__ ckv_full, bf16* __restrict__ ckv,
                    bf16* __restrict__ kpe) {
  int row = blockIdx.x;            // b*S + s
  int s = row % kS;
  const float* cr = ckv_full + (long)row * (kKVLR + kHD);
  for (int e = threadIdx.x; e < kKVLR; e += 256)
    ckv[(long)row * kKVLR + e] = (bf16)cr[e];
  for (int d = threadIdx.x; d < kHD; d += 256) {
    float x = cr[kKVLR + d];
    int dm = d & 63;
    float invf = powf(10000.0f, -(float)(2 * dm) / 128.0f);
    float ang = (float)s * invf;
    float x2 = (d < 64) ? -cr[kKVLR + d + 64] : cr[kKVLR + d - 64];
    kpe[(long)row * kHD + d] = (bf16)(x * cosf(ang) + x2 * sinf(ang));
  }
}

// ---------------------------------------------------------------------------
// Fused flash-MLA core. Grid (S/32, NH, B), 128 threads (4 waves).
// Double-buffered K tiles staged by TDM (per-wave 1-D descriptors) or
// global_load_async_to_lds, overlapped with WMMA compute.
// ---------------------------------------------------------------------------
__global__ __launch_bounds__(128)
void mla_fa_kernel(const bf16* __restrict__ q_lat, const bf16* __restrict__ q_pe,
                   const bf16* __restrict__ ckv,   const bf16* __restrict__ k_pe,
                   const float* __restrict__ mask, bf16* __restrict__ out_lat,
                   float scale) {
  __shared__ __align__(16) bf16 sQ [32 * 512];
  __shared__ __align__(16) bf16 sQp[32 * 128];
  __shared__ __align__(16) bf16 sK [2][32 * 512];
  __shared__ __align__(16) bf16 sKp[2][32 * 128];
  __shared__ float sS[32 * 32];
  __shared__ __align__(16) bf16 sP[32 * 32];
  __shared__ float sM[32], sL[32], sC[32];

  int s0 = blockIdx.x * 32;
  int h  = blockIdx.y;
  int b  = blockIdx.z;
  int tid = threadIdx.x, lane = tid & 31, w = tid >> 5;

  const bf16* qb  = q_lat + ((long)(b * kNH + h) * kS + s0) * kKVLR;
  const bf16* qpb = q_pe  + ((long)(b * kNH + h) * kS + s0) * kHD;

  // ---- staging helpers -----------------------------------------------------
  auto stage_q = [&]() {
#if USE_TDM
    // contiguous per-wave quarters: 2 descriptors per wave
    tdm_copy_1d(qb  + (long)w * 8 * kKVLR, &sQ [w * 8 * kKVLR], 8 * kKVLR);
    tdm_copy_1d(qpb + (long)w * 8 * kHD,   &sQp[w * 8 * kHD],   8 * kHD);
#else
    for (int i = tid; i < 32 * 512 / 8; i += 128) {
      int r = i >> 6, c = (i & 63) * 8;
      g2l_b128(qb + (long)r * 512 + c, &sQ[r * 512 + c]);
    }
    for (int i = tid; i < 32 * 128 / 8; i += 128) {
      int r = i >> 4, c = (i & 15) * 8;
      g2l_b128(qpb + (long)r * 128 + c, &sQp[r * 128 + c]);
    }
#endif
  };
  auto stage_k = [&](int t0, int bsel) {
    const bf16* kb  = ckv  + ((long)b * kS + t0) * kKVLR;
    const bf16* kpb = k_pe + ((long)b * kS + t0) * kHD;
#if USE_TDM
    tdm_copy_1d(kb  + (long)w * 8 * kKVLR, &sK [bsel][w * 8 * kKVLR], 8 * kKVLR);
    tdm_copy_1d(kpb + (long)w * 8 * kHD,   &sKp[bsel][w * 8 * kHD],   8 * kHD);
#else
    for (int i = tid; i < 32 * 512 / 8; i += 128) {
      int r = i >> 6, c = (i & 63) * 8;
      g2l_b128(kb + (long)r * 512 + c, &sK[bsel][r * 512 + c]);
    }
    for (int i = tid; i < 32 * 128 / 8; i += 128) {
      int r = i >> 4, c = (i & 15) * 8;
      g2l_b128(kpb + (long)r * 128 + c, &sKp[bsel][r * 128 + c]);
    }
#endif
  };

  stage_q();
  stage_k(0, 0);
  if (tid < 32) { sM[tid] = -__builtin_inff(); sL[tid] = 0.f; }

  v8f acc[2][8];
#pragma unroll
  for (int i = 0; i < 2; ++i)
#pragma unroll
    for (int j = 0; j < 8; ++j)
#pragma unroll
      for (int v = 0; v < 8; ++v) acc[i][j][v] = 0.f;

  int buf = 0;
  for (int t0 = 0; t0 < kS; t0 += 32) {
    bool hn = (t0 + 32 < kS);
    if (hn) stage_k(t0 + 32, buf ^ 1);
    // Counted waits: leave only the prefetched next tile in flight.
    // TDM: 2 descriptors/wave/tile. ASYNC: 20 ops/thread/tile.
#if USE_TDM
    if (hn) TENSOR_WAIT(2); else TENSOR_WAIT(0);
#else
    if (hn) ASYNC_WAIT(20); else ASYNC_WAIT(0);
#endif
    __syncthreads();

    const bf16* sKb  = &sK [buf][0];
    const bf16* sKpb = &sKp[buf][0];

    {   // score tile: wave w -> (rb = w&1, cb = w>>1), K-reduction 640
      int rb = w & 1, cb = w >> 1;
      v8f sc;
#pragma unroll
      for (int v = 0; v < 8; ++v) sc[v] = 0.f;
#pragma unroll
      for (int kk = 0; kk < 512; kk += 32)
        sc = wmma_bf16(lds_a_frag(sQ,  rb * 16, 512, kk),
                       lds_b_frag(sKb, cb * 16, 512, kk), sc);
#pragma unroll
      for (int kk = 0; kk < 128; kk += 32)
        sc = wmma_bf16(lds_a_frag(sQp,  rb * 16, 128, kk),
                       lds_b_frag(sKpb, cb * 16, 128, kk), sc);
      int g = lane >> 4, n = lane & 15;
#pragma unroll
      for (int v = 0; v < 8; ++v)
        sS[(rb * 16 + v + 8 * g) * 32 + cb * 16 + n] = sc[v];
    }
    __syncthreads();

    if (tid < 32) {   // per-row online softmax stats
      int r = tid;
      const float* mrow = mask + ((long)b * kS + (s0 + r)) * kS + t0;
      float vals[32];
      float mx = -__builtin_inff();
#pragma unroll
      for (int j = 0; j < 32; ++j) {
        float vv = sS[r * 32 + j] * scale + mrow[j];
        vals[j] = vv;
        mx = fmaxf(mx, vv);
      }
      float m_old = sM[r];
      float m_new = fmaxf(m_old, mx);
      float corr  = __expf(m_old - m_new);
      float lsum  = 0.f;
#pragma unroll
      for (int j = 0; j < 32; ++j) {
        float p = __expf(vals[j] - m_new);
        lsum += p;
        sP[r * 32 + j] = (bf16)p;
      }
      sC[r] = corr; sM[r] = m_new;
      sL[r] = sL[r] * corr + lsum;
    }
    __syncthreads();

    {   // acc = acc*corr + P @ ckv_tile ; wave w -> cols [w*128, w*128+128)
      int g = lane >> 4;
#pragma unroll
      for (int rb = 0; rb < 2; ++rb) {
        v16bf af = lds_a_frag(sP, rb * 16, 32, 0);
#pragma unroll
        for (int cb = 0; cb < 8; ++cb) {
#pragma unroll
          for (int v = 0; v < 8; ++v)
            acc[rb][cb][v] *= sC[rb * 16 + v + 8 * g];
          v16bf bfr = lds_bt_frag(sKb, 0, w * 128 + cb * 16, 512);
          acc[rb][cb] = wmma_bf16(af, bfr, acc[rb][cb]);
        }
      }
    }
    __syncthreads();
    buf ^= 1;
  }

  bf16* ob = out_lat + ((long)(b * kNH + h) * kS + s0) * kKVLR;
  int g = lane >> 4, n = lane & 15;
#pragma unroll
  for (int rb = 0; rb < 2; ++rb)
#pragma unroll
    for (int cb = 0; cb < 8; ++cb)
#pragma unroll
      for (int v = 0; v < 8; ++v) {
        int row = rb * 16 + v + 8 * g;
        int col = w * 128 + cb * 16 + n;
        ob[(long)row * kKVLR + col] = (bf16)(acc[rb][cb][v] / sL[row]);
      }
}

// ---------------------------------------------------------------------------
// Host-side launcher
// ---------------------------------------------------------------------------
extern "C" void kernel_launch(void* const* d_in, const int* in_sizes, int n_in,
                              void* d_out, int out_size, void* d_ws, size_t ws_size,
                              hipStream_t stream) {
  (void)in_sizes; (void)n_in; (void)out_size; (void)ws_size;

  const float* hidden     = (const float*)d_in[0];
  const float* attn_mask  = (const float*)d_in[1];
  const float* q_a_W      = (const float*)d_in[2];
  const float* q_a_b      = (const float*)d_in[3];
  const float* q_a_norm_w = (const float*)d_in[4];
  const float* q_b_W      = (const float*)d_in[5];
  const float* kv_a_W     = (const float*)d_in[6];
  const float* kv_b_W     = (const float*)d_in[7];
  const float* o_W        = (const float*)d_in[8];
  float* out = (float*)d_out;

  const long MTOK = (long)kB * kS;   // 4096 token rows

  char* p = (char*)d_ws;
  size_t cur = 0;
  auto take = [&](size_t bytes) -> char* {
    char* r = p + cur;
    cur += (bytes + 255) & ~(size_t)255;
    return r;
  };
  bf16* hidden_bf = (bf16*)take(MTOK * kHID * 2);
  bf16* qaW_bf    = (bf16*)take((long)kQLR * kHID * 2);
  bf16* qbW_bf    = (bf16*)take((long)kNH * kQHD * kQLR * 2);
  bf16* kvaW_bf   = (bf16*)take((long)(kKVLR + kHD) * kHID * 2);
  bf16* kvbW_bf   = (bf16*)take((long)kKVLR * kNH * 2 * kHD * 2);
  bf16* oW_bf     = (bf16*)take((long)kHID * kNH * kHD * 2);
  bf16* qa_n      = (bf16*)take(MTOK * kQLR * 2);
  bf16* q_nope    = (bf16*)take((long)kB * kNH * kS * kHD * 2);
  bf16* q_pe      = (bf16*)take((long)kB * kNH * kS * kHD * 2);
  bf16* ckv_bf    = (bf16*)take(MTOK * kKVLR * 2);
  bf16* kpe_bf    = (bf16*)take(MTOK * kHD * 2);
  bf16* attn_out  = (bf16*)take(MTOK * kNH * kHD * 2);
  // scratch union region (f32 temps die before the bf16 users are written)
  char* scr = take(102760448);
  float* qa_f     = (float*)(scr);
  float* q_full   = (float*)(scr + 25165824);
  float* ckv_full = (float*)(scr + 92274688);
  bf16*  q_lat    = (bf16*)(scr);
  bf16*  attn_lat = (bf16*)(scr + 33554432);

  // 1. casts
  cast_f32_bf16_kernel<<<1024, 256, 0, stream>>>(hidden, hidden_bf, MTOK * kHID);
  cast_f32_bf16_kernel<<<1024, 256, 0, stream>>>(q_a_W, qaW_bf, (long)kQLR * kHID);
  cast_f32_bf16_kernel<<<1024, 256, 0, stream>>>(q_b_W, qbW_bf, (long)kNH * kQHD * kQLR);
  cast_f32_bf16_kernel<<<1024, 256, 0, stream>>>(kv_a_W, kvaW_bf, (long)(kKVLR + kHD) * kHID);
  cast_f32_bf16_kernel<<<1024, 256, 0, stream>>>(kv_b_W, kvbW_bf, (long)kKVLR * kNH * 2 * kHD);
  cast_f32_bf16_kernel<<<1024, 256, 0, stream>>>(o_W, oW_bf, (long)kHID * kNH * kHD);

  // 2. q_a = hidden @ q_a_W^T + b
  gemm_bf16_kernel<<<dim3(kQLR / 128, MTOK / 128, 1), 256, 0, stream>>>(
      hidden_bf, qaW_bf, q_a_b, qa_f, nullptr,
      (int)MTOK, kQLR, kHID, kHID, kHID, 1, kQLR, 1, 0, 0, 0, 0, 0, 0);

  // 3. RMSNorm -> bf16
  rmsnorm_kernel<<<(int)MTOK, 256, 0, stream>>>(qa_f, q_a_norm_w, qa_n, kQLR);

  // 4. q = qa_n @ q_b_W^T
  gemm_bf16_kernel<<<dim3((kNH * kQHD) / 128, MTOK / 128, 1), 256, 0, stream>>>(
      qa_n, qbW_bf, nullptr, q_full, nullptr,
      (int)MTOK, kNH * kQHD, kQLR, kQLR, kQLR, 1, kNH * kQHD, 1, 0, 0, 0, 0, 0, 0);

  // 5. ckv_full = hidden @ kv_a_W^T
  gemm_bf16_kernel<<<dim3((kKVLR + kHD) / 128, MTOK / 128, 1), 256, 0, stream>>>(
      hidden_bf, kvaW_bf, nullptr, ckv_full, nullptr,
      (int)MTOK, kKVLR + kHD, kHID, kHID, kHID, 1, kKVLR + kHD, 1, 0, 0, 0, 0, 0, 0);

  // 6/7. RoPE + splits
  rope_q_kernel<<<(int)MTOK, 256, 0, stream>>>(q_full, q_nope, q_pe);
  rope_kv_kernel<<<(int)MTOK, 256, 0, stream>>>(ckv_full, ckv_bf, kpe_bf);

  // 8. q_lat[b,h] = q_nope[b,h] @ q_absorb[h]^T  (Z = B*NH)
  gemm_bf16_kernel<<<dim3(kKVLR / 128, kS / 128, kB * kNH), 256, 0, stream>>>(
      q_nope, kvbW_bf, nullptr, nullptr, q_lat,
      kS, kKVLR, kHD, kHD, (long)kNH * 2 * kHD, 1, kKVLR,
      kNH,
      (long)kNH * kS * kHD, (long)kS * kHD,
      0, (long)2 * kHD,
      (long)kNH * kS * kKVLR, (long)kS * kKVLR);

  // 9. fused flash-MLA attention
  float scale = 1.0f / sqrtf((float)(kKVLR + kHD));
  mla_fa_kernel<<<dim3(kS / 32, kNH, kB), 128, 0, stream>>>(
      q_lat, q_pe, ckv_bf, kpe_bf, attn_mask, attn_lat, scale);

  // 10. attn_out[b,s,h*128+d] = attn_lat[b,h] @ o_absorb[h]
  gemm_bf16_kernel<<<dim3(kHD / 128, kS / 128, kB * kNH), 256, 0, stream>>>(
      attn_lat, kvbW_bf + kHD, nullptr, nullptr, attn_out,
      kS, kHD, kKVLR, kKVLR, 1, (long)kNH * 2 * kHD, (long)kNH * kHD,
      kNH,
      (long)kNH * kS * kKVLR, (long)kS * kKVLR,
      0, (long)2 * kHD,
      (long)kS * kNH * kHD, (long)kHD);

  // 11. out = attn_out @ o_W^T  -> f32 d_out
  gemm_bf16_kernel<<<dim3(kHID / 128, MTOK / 128, 1), 256, 0, stream>>>(
      attn_out, oW_bf, nullptr, out, nullptr,
      (int)MTOK, kHID, kNH * kHD, (long)kNH * kHD, (long)kNH * kHD, 1, kHID,
      1, 0, 0, 0, 0, 0, 0);
}
